// TSTRefiner_6579889897595
// MI455X (gfx1250) — matrix operand order repaired and checked by reading, not compiled, not run on hardware
//
#include <hip/hip_runtime.h>
#include <hip/hip_fp16.h>

// ---------------------------------------------------------------------------
// MI455X (gfx1250, wave32) TST refiner.
// All matmuls via v_wmma_f32_16x16x32_f16. Banded cross-attention uses a
// flash-style online softmax over the contiguous segment band (~300 frames
// per query instead of 40000). Scores are computed TRANSPOSED (frames=M,
// queries=N) so softmax reductions are per-lane: only 2 cross-lane permutes
// per chunk (wave32 shuffles are ds_bpermute on gfx1250 -> keep them rare).
// The 64MB mask-logits output is written with non-temporal stores so it does
// not displace L2-resident memory/pos/frame_feat.
// ---------------------------------------------------------------------------

typedef __attribute__((ext_vector_type(16))) _Float16 v16h;
typedef __attribute__((ext_vector_type(8)))  float    v8f;

#define H_DIM 512
#define FF_DIM 512
#define S_Q   400
#define T_F   40000
#define INNER_D 64
#define NCp1  49

__device__ __forceinline__ v8f wmma_f16(v16h a, v16h b, v8f c) {
  // D = A(16x32 f16) * B(32x16 f16) + C(16x16 f32)
  return __builtin_amdgcn_wmma_f32_16x16x32_f16(
      /*neg_a=*/false, a, /*neg_b=*/false, b,
      /*c_mod=*/(short)0, c, /*reuse_a=*/false, /*reuse_b=*/false);
}

// ---------------------------------------------------------------------------
// Segment scan: seg_id[t] = #label-changes <= t; per-query band [lo,hi).
// ---------------------------------------------------------------------------
__global__ __launch_bounds__(256)
void seg_kernel(const int* __restrict__ act, int Tdim, int Sq,
                int* __restrict__ seg_id, int* __restrict__ seg_start,
                int* __restrict__ lo, int* __restrict__ hi)
{
  __shared__ int sc[256];
  __shared__ int offs;
  const int tid = threadIdx.x;
  if (tid == 0) { offs = 0; seg_start[0] = 0; }
  __syncthreads();
  for (int base = 0; base < Tdim; base += 256) {
    int t = base + tid;
    int chg = 0;
    if (t > 0 && t < Tdim) chg = (act[t] != act[t - 1]) ? 1 : 0;
    sc[tid] = chg;
    __syncthreads();
    for (int d = 1; d < 256; d <<= 1) {
      int v = (tid >= d) ? sc[tid - d] : 0;
      __syncthreads();
      sc[tid] += v;
      __syncthreads();
    }
    if (t < Tdim) {
      int sid = offs + sc[tid];
      seg_id[t] = sid;
      if (chg) seg_start[sid] = t;
    }
    __syncthreads();
    if (tid == 0) offs += sc[255];
    __syncthreads();
  }
  const int nseg = offs + 1;
  if (tid == 0) seg_start[nseg] = Tdim;
  __syncthreads();
  for (int s = tid; s < Sq; s += 256) {
    int glo = (s - 1 < 0) ? 0 : s - 1;
    int ghi = (s + 1 > nseg - 1) ? nseg - 1 : s + 1;
    if (glo > nseg - 1) { lo[s] = 0; hi[s] = 0; }
    else { lo[s] = seg_start[glo]; hi[s] = seg_start[ghi + 1]; }
  }
}

// ---------------------------------------------------------------------------
// Flash attention, transposed-score formulation.
// Block = 128 threads = 4 wave32; 16 queries per block; each wave owns a
// 128-wide slice of H. Per 16-frame chunk and wave:
//   S^T = K_tile x Q^T : 16 x v_wmma (fully unrolled K=512)
//   O^T += V^T x P^T   :  8 x v_wmma (frame-K padded 16->32 with reg zeros)
// Shared memory is manually carved and the 32KB q/k region is reused as the
// f32 transpose buffer for coalesced output stores.
// ---------------------------------------------------------------------------
__global__ __launch_bounds__(128)
void flash_attn_kernel(const float* __restrict__ qsrc, const float* __restrict__ qpos,
                       const float* __restrict__ ksrc, const float* __restrict__ kpos,
                       const float* __restrict__ vsrc,
                       const int* __restrict__ lo_arr, const int* __restrict__ hi_arr,
                       const int* __restrict__ seg_id, int use_mask,
                       int Tdim, float* __restrict__ outp)
{
  __shared__ __align__(16) char smem[51328];
  _Float16 (*q_h)[H_DIM]  = reinterpret_cast<_Float16(*)[H_DIM]>(smem);          // 16 KB
  _Float16 (*k_h)[H_DIM]  = reinterpret_cast<_Float16(*)[H_DIM]>(smem + 16384);  // 16 KB
  _Float16 (*v_h)[16]     = reinterpret_cast<_Float16(*)[16]>(smem + 32768);     // 16 KB transposed V
  _Float16 (*p_h)[16][16] = reinterpret_cast<_Float16(*)[16][16]>(smem + 49152); // 2 KB
  int* seg_sh  = reinterpret_cast<int*>(smem + 51200);
  int* band_sh = reinterpret_cast<int*>(smem + 51264);

  const int tid  = threadIdx.x;
  const int wave = tid >> 5;
  const int lane = tid & 31;
  const int hl   = lane & 15;
  const int hi16 = lane >> 4;
  const int s0   = blockIdx.x * 16;
  const int q    = s0 + hl;              // this lane's query column

  for (int i = tid; i < 16 * H_DIM; i += 128) {
    int r = i >> 9, h = i & (H_DIM - 1);
    q_h[r][h] = (_Float16)(qsrc[(s0 + r) * H_DIM + h] + qpos[(s0 + r) * H_DIM + h]);
  }
  if (tid == 0) {
    int blo = Tdim, bhi = 0;
    for (int r = 0; r < 16; ++r) {
      int l = use_mask ? lo_arr[s0 + r] : 0;
      int h = use_mask ? hi_arr[s0 + r] : Tdim;
      blo = (l < blo) ? l : blo;
      bhi = (h > bhi) ? h : bhi;
    }
    band_sh[0] = blo; band_sh[1] = bhi;
  }
  __syncthreads();
  const int blo = band_sh[0], bhi = band_sh[1];

  float Mq = -3.0e38f, Lq = 0.f;     // per-query (per-lane) softmax state
  v8f O[8];
  #pragma unroll
  for (int o = 0; o < 8; ++o) O[o] = (v8f){0.f,0.f,0.f,0.f,0.f,0.f,0.f,0.f};
  const float scale = 0.044194173824159216f; // 1/sqrt(512)

  for (int t0 = blo; t0 < bhi; t0 += 16) {
    __syncthreads(); // previous chunk fully consumed before LDS reuse
    for (int i = tid; i < 16 * H_DIM; i += 128) {
      int tl = i >> 9, h = i & (H_DIM - 1);
      int t = t0 + tl;
      float kv = 0.f, vv = 0.f;
      if (t < bhi) {
        kv = ksrc[t * H_DIM + h] + kpos[t * H_DIM + h];
        vv = vsrc[t * H_DIM + h];
        if (t + 16 < Tdim) __builtin_prefetch(&ksrc[(t + 16) * H_DIM + h], 0, 1);
      }
      k_h[tl][h] = (_Float16)kv;
      v_h[h][tl] = (_Float16)vv;
    }
    if (tid < 16) {
      int t = t0 + tid;
      seg_sh[tid] = (use_mask && t < bhi) ? seg_id[t] : 0;
    }
    __syncthreads();

    // ---- S^T [16 frames x 16 queries], K = 512, fully unrolled ----
    v8f acc = (v8f){0.f,0.f,0.f,0.f,0.f,0.f,0.f,0.f};
    #pragma unroll
    for (int kk = 0; kk < H_DIM; kk += 32) {
      v16h a, b;
      const int aoff = kk + hi16 * 8;       // A = K tile, row = frame
      #pragma unroll
      for (int i = 0; i < 8; ++i) {
        a[i]     = k_h[hl][aoff + i];
        a[8 + i] = k_h[hl][aoff + 16 + i];
      }
      const int boff = kk + hi16 * 8;       // B = Q^T, col = query
      #pragma unroll
      for (int i = 0; i < 16; ++i) b[i] = q_h[hl][boff + i + (i & 8)];
      acc = wmma_f16(a, b, acc);
    }

    // ---- per-lane online softmax: 2 cross-lane ops per chunk ----
    float x[8];
    float mloc = -3.0e38f;
    #pragma unroll
    for (int e = 0; e < 8; ++e) {
      const int fr = e + 8 * hi16;          // frame row held by this lane
      const int dd = seg_sh[fr] - q;
      const bool ok = (t0 + fr < bhi) && (!use_mask || (dd >= -1 && dd <= 1));
      x[e] = ok ? acc[e] * scale : -3.0e38f;
      mloc = fmaxf(mloc, x[e]);
    }
    mloc = fmaxf(mloc, __shfl_xor(mloc, 16, 32));
    const float Mn  = fmaxf(Mq, mloc);
    const bool  any = (Mn > -1.0e37f);
    const float al  = any ? __expf(Mq - Mn) : 1.f;
    float rs = 0.f;
    #pragma unroll
    for (int e = 0; e < 8; ++e) {
      const float p = any ? __expf(x[e] - Mn) : 0.f;
      rs += p;
      p_h[wave][e + 8 * hi16][hl] = (_Float16)p;   // P^T staging
    }
    rs += __shfl_xor(rs, 16, 32);
    Lq = Lq * al + rs;
    Mq = Mn;
    #pragma unroll
    for (int o = 0; o < 8; ++o) {
      #pragma unroll
      for (int e = 0; e < 8; ++e) O[o][e] *= al;   // v_pk_mul_f32 broadcast
    }
    // same-wave LDS producer->consumer: DS is in-order; fence the compiler
    asm volatile("s_wait_dscnt 0x0" ::: "memory");

    // ---- O^T += V^T(16h x 32f; frames 16..31 = reg zeros) x P^T(32f x 16q)
    v16h pb;
    #pragma unroll
    for (int i = 0; i < 16; ++i)
      pb[i] = (i >= 8) ? (_Float16)0.f : p_h[wave][hi16 * 8 + i][hl];
    #pragma unroll
    for (int ot = 0; ot < 8; ++ot) {
      const int hrow = wave * 128 + ot * 16 + hl;
      v16h va;
      #pragma unroll
      for (int i = 0; i < 8; ++i) {
        va[i]     = v_h[hrow][hi16 * 8 + i];
        va[8 + i] = (_Float16)0.f;
      }
      O[ot] = wmma_f16(va, pb, O[ot]);
    }
  }

  // normalize + coalesced store via LDS transpose (reuse q/k 32KB region)
  const float invq = (Lq > 0.f) ? 1.f / Lq : 0.f;
  __syncthreads(); // all waves done with q_h/k_h
  float* obuf = reinterpret_cast<float*>(smem) + wave * 2048; // 8KB per wave
  #pragma unroll
  for (int ot = 0; ot < 8; ++ot) {
    #pragma unroll
    for (int e = 0; e < 8; ++e)
      obuf[hl * 128 + ot * 16 + e + 8 * hi16] = O[ot][e] * invq;
  }
  asm volatile("s_wait_dscnt 0x0" ::: "memory");
  for (int i = lane; i < 2048; i += 32) {
    int qq = i >> 7, hloc = i & 127;
    outp[(s0 + qq) * H_DIM + wave * 128 + hloc] = obuf[i];
  }
}

// ---------------------------------------------------------------------------
// WMMA GEMM: 4 waves/block, one 16x16 tile per wave (consecutive N tiles).
// C = act(A) @ B (+bias)(+resid); b_nmajor: B stored [N,K] row-major.
// nt_out: stream the result with non-temporal stores (TH=NT) for outputs
// that have no on-device consumer (mask_logits, 64MB).
// ---------------------------------------------------------------------------
__global__ __launch_bounds__(128)
void gemm16_kernel(const float* __restrict__ A, const float* __restrict__ B,
                   const float* __restrict__ bias, const float* __restrict__ resid,
                   float* __restrict__ C,
                   int Mm, int Nn, int Kk, int relu_in, int relu_out,
                   int b_nmajor, int nt_out)
{
  const int wave = threadIdx.x >> 5;
  const int lane = threadIdx.x & 31;
  const int hl   = lane & 15;
  const int hi16 = lane >> 4;
  const int m0 = blockIdx.x * 16;
  const int n0 = (blockIdx.y * 4 + wave) * 16;
  const int arow = m0 + hl;
  const int bcol = n0 + hl;
  v8f acc = (v8f){0.f,0.f,0.f,0.f,0.f,0.f,0.f,0.f};

  #pragma unroll 4
  for (int k = 0; k < Kk; k += 32) {
    v16h a, b;
    const int aoff = k + hi16 * 8;
    #pragma unroll
    for (int i = 0; i < 8; ++i) {
      float x = (arow < Mm) ? A[arow * Kk + aoff + i] : 0.f;
      float y = (arow < Mm) ? A[arow * Kk + aoff + 16 + i] : 0.f;
      if (relu_in) { x = fmaxf(x, 0.f); y = fmaxf(y, 0.f); }
      a[i] = (_Float16)x; a[8 + i] = (_Float16)y;
    }
    const int koff = k + hi16 * 8;
    #pragma unroll
    for (int i = 0; i < 16; ++i) {
      float x = 0.f;
      if (bcol < Nn)
        x = b_nmajor ? B[bcol * Kk + koff + i + (i & 8)] : B[(koff + i + (i & 8)) * Nn + bcol];
      b[i] = (_Float16)x;
    }
    acc = wmma_f16(a, b, acc);
  }
  #pragma unroll
  for (int e = 0; e < 8; ++e) {
    const int row = m0 + e + hi16 * 8;
    const int col = n0 + hl;
    if (row < Mm && col < Nn) {
      float v = acc[e];
      if (bias)  v += bias[col];
      if (resid) v += resid[row * Nn + col];
      if (relu_out) v = fmaxf(v, 0.f);
      if (nt_out) __builtin_nontemporal_store(v, &C[row * Nn + col]);
      else        C[row * Nn + col] = v;
    }
  }
}

// ---------------------------------------------------------------------------
// out = LayerNorm(x + y) per row of H_DIM=512; 256 threads, 2 elems/thread.
// ---------------------------------------------------------------------------
__global__ __launch_bounds__(256)
void add_ln_kernel(const float* __restrict__ x, const float* __restrict__ y,
                   const float* __restrict__ g, const float* __restrict__ b,
                   float* __restrict__ out)
{
  __shared__ float rbuf[8];
  __shared__ float stat[2];
  const int row = blockIdx.x;
  const int tid = threadIdx.x;
  const int wid = tid >> 5;

  const float v0 = x[row * H_DIM + tid]       + y[row * H_DIM + tid];
  const float v1 = x[row * H_DIM + 256 + tid] + y[row * H_DIM + 256 + tid];

  float s = v0 + v1;
  #pragma unroll
  for (int d = 1; d < 32; d <<= 1) s += __shfl_xor(s, d, 32);
  if ((tid & 31) == 0) rbuf[wid] = s;
  __syncthreads();
  if (tid == 0) { float t = 0.f; for (int i = 0; i < 8; ++i) t += rbuf[i]; stat[0] = t / (float)H_DIM; }
  __syncthreads();
  const float mu = stat[0];
  const float d0 = v0 - mu, d1 = v1 - mu;
  float s2 = d0 * d0 + d1 * d1;
  #pragma unroll
  for (int d = 1; d < 32; d <<= 1) s2 += __shfl_xor(s2, d, 32);
  if ((tid & 31) == 0) rbuf[wid] = s2;
  __syncthreads();
  if (tid == 0) { float t = 0.f; for (int i = 0; i < 8; ++i) t += rbuf[i]; stat[1] = t / (float)H_DIM; }
  __syncthreads();
  const float rstd = rsqrtf(stat[1] + 1e-5f);
  out[row * H_DIM + tid]       = g[tid] * d0 * rstd + b[tid];
  out[row * H_DIM + 256 + tid] = g[tid + 256] * d1 * rstd + b[tid + 256];
}

// ---------------------------------------------------------------------------
// Host-side orchestration
// ---------------------------------------------------------------------------
extern "C" void kernel_launch(void* const* d_in, const int* in_sizes, int n_in,
                              void* d_out, int out_size, void* d_ws, size_t ws_size,
                              hipStream_t stream) {
  (void)in_sizes; (void)n_in; (void)out_size; (void)ws_size;

  const float* tgt0   = (const float*)d_in[0];  // [S,1,H]
  const float* memory = (const float*)d_in[1];  // [T,1,H]
  const float* pos    = (const float*)d_in[2];  // [T,1,H]
  const float* qpos   = (const float*)d_in[3];  // [S,1,H]
  const int*   action = (const int*)d_in[4];    // [T]
  const float* ffeat  = (const float*)d_in[5];  // [1,T,INNER]
  // params flattened in dict order: ca1(14), sa(14), ca2(14), cls_w, cls_b, mask_w, mask_b
  auto Pp = [&](int i) { return (const float*)d_in[6 + i]; };
  // per-block offsets: 0 tw, 1 tb, 2 cw, 3 cb, 4 w1, 5 b1, 6 w2, 7 b2,
  //                    8 ln1_g, 9 ln1_b, 10 ln2_g, 11 ln2_b, 12 ln3_g, 13 ln3_b

  // workspace carve
  char* w = (char*)d_ws;
  int* seg_id    = (int*)w;  w += (size_t)T_F * sizeof(int);
  int* seg_start = (int*)w;  w += (size_t)(T_F + 2) * sizeof(int);
  int* lo        = (int*)w;  w += (size_t)S_Q * sizeof(int);
  int* hi        = (int*)w;  w += (size_t)S_Q * sizeof(int);
  float* attn  = (float*)w;  w += (size_t)S_Q * H_DIM * sizeof(float);
  float* lin   = (float*)w;  w += (size_t)S_Q * H_DIM * sizeof(float);
  float* t1    = (float*)w;  w += (size_t)S_Q * H_DIM * sizeof(float);
  float* f1    = (float*)w;  w += (size_t)S_Q * FF_DIM * sizeof(float);
  float* f2    = (float*)w;  w += (size_t)S_Q * H_DIM * sizeof(float);
  float* tgtA  = (float*)w;  w += (size_t)S_Q * H_DIM * sizeof(float);
  float* tgtB  = (float*)w;  w += (size_t)S_Q * H_DIM * sizeof(float);
  float* membd = (float*)w;  w += (size_t)S_Q * INNER_D * sizeof(float);

  seg_kernel<<<1, 256, 0, stream>>>(action, T_F, S_Q, seg_id, seg_start, lo, hi);

  const dim3 gHH(S_Q / 16, (H_DIM / 16) / 4);   // (25, 8)
  const dim3 gHF(S_Q / 16, (FF_DIM / 16) / 4);

  auto cross = [&](const float* tin, float* tout, int pb) {
    flash_attn_kernel<<<S_Q / 16, 128, 0, stream>>>(
        tin, qpos, memory, pos, memory, lo, hi, seg_id, /*use_mask=*/1, T_F, attn);
    // tgt2 = relu(attn) @ tw + tb
    gemm16_kernel<<<gHH, 128, 0, stream>>>(attn, Pp(pb + 0), Pp(pb + 1), nullptr, lin,
                                           S_Q, H_DIM, H_DIM, 1, 0, 0, 0);
    add_ln_kernel<<<S_Q, 256, 0, stream>>>(tin, lin, Pp(pb + 10), Pp(pb + 11), t1);
    gemm16_kernel<<<gHF, 128, 0, stream>>>(t1, Pp(pb + 4), Pp(pb + 5), nullptr, f1,
                                           S_Q, FF_DIM, H_DIM, 0, 1, 0, 0);
    gemm16_kernel<<<gHH, 128, 0, stream>>>(f1, Pp(pb + 6), Pp(pb + 7), nullptr, f2,
                                           S_Q, H_DIM, FF_DIM, 0, 0, 0, 0);
    add_ln_kernel<<<S_Q, 256, 0, stream>>>(t1, f2, Pp(pb + 12), Pp(pb + 13), tout);
  };
  auto selfl = [&](const float* tin, float* tout, int pb) {
    flash_attn_kernel<<<S_Q / 16, 128, 0, stream>>>(
        tin, qpos, tin, qpos, tin, lo, hi, seg_id, /*use_mask=*/0, S_Q, attn);
    // src2 = relu(attn) @ cw + cb
    gemm16_kernel<<<gHH, 128, 0, stream>>>(attn, Pp(pb + 2), Pp(pb + 3), nullptr, lin,
                                           S_Q, H_DIM, H_DIM, 1, 0, 0, 0);
    add_ln_kernel<<<S_Q, 256, 0, stream>>>(tin, lin, Pp(pb + 8), Pp(pb + 9), t1);
    gemm16_kernel<<<gHF, 128, 0, stream>>>(t1, Pp(pb + 4), Pp(pb + 5), nullptr, f1,
                                           S_Q, FF_DIM, H_DIM, 0, 1, 0, 0);
    gemm16_kernel<<<gHH, 128, 0, stream>>>(f1, Pp(pb + 6), Pp(pb + 7), nullptr, f2,
                                           S_Q, H_DIM, FF_DIM, 0, 0, 0, 0);
    add_ln_kernel<<<S_Q, 256, 0, stream>>>(t1, f2, Pp(pb + 10), Pp(pb + 11), tout);
  };

  cross(tgt0, tgtA, 0);    // ca1
  selfl(tgtA, tgtB, 14);   // sa (B=1: transposes are layout no-ops)
  cross(tgtB, tgtA, 28);   // ca2

  // heads
  float* cls_out = (float*)d_out;                 // [S, NC+1]
  float* mlog    = (float*)d_out + S_Q * NCp1;    // [S, T]
  gemm16_kernel<<<dim3(S_Q / 16, 1), 128, 0, stream>>>(
      tgtA, Pp(42), Pp(43), nullptr, cls_out, S_Q, NCp1, H_DIM, 0, 0, 0, 0);
  gemm16_kernel<<<dim3(S_Q / 16, 1), 128, 0, stream>>>(
      tgtA, Pp(44), Pp(45), nullptr, membd, S_Q, INNER_D, H_DIM, 0, 0, 0, 0);
  // mask_logits = mask_embed @ frame_feat^T (B stored [T,INNER] row-major),
  // streamed to memory with non-temporal stores (no on-device consumer).
  gemm16_kernel<<<dim3(S_Q / 16, (T_F / 16) / 4), 128, 0, stream>>>(
      membd, ffeat, nullptr, nullptr, mlog, S_Q, T_F, INNER_D, 0, 0, 1, 1);
}
